// MMOE_33002528703207
// MI455X (gfx1250) — matrix operand order repaired and checked by reading, not compiled, not run on hardware
//
#include <hip/hip_runtime.h>
#include <hip/hip_bf16.h>

typedef __attribute__((ext_vector_type(16))) _Float16 half16;
typedef __attribute__((ext_vector_type(8)))  _Float16 half8;
typedef __attribute__((ext_vector_type(8)))  float    float8;

#define BATCH 65536
#define NS 26
#define ND 13
#define EMBD 5
#define DIN 143
#define KPAD 160
#define H1 128
#define H2 64
#define NEXP 4
#define ROWS2 32              // rows per block (2 tiles of 16)
#define VOCAB 100000

// Pre-swizzled f16 weight fragment layout in d_ws:
//   W1: 4 experts x 5 K-steps x 8 N-tiles = 160 fragments
//   W2: 4 experts x 4 K-steps x 4 N-tiles =  64 fragments
// Each fragment = 32 lanes x 16 halves (one half16 per lane, contiguous 32B).
#define W1FRAGS 160
#define W2FRAGS 64
#define FRAGH   512                      // halves per fragment
#define W1H     (W1FRAGS * FRAGH)        // 81920 halves
#define W2H     (W2FRAGS * FRAGH)        // 32768 halves
#define WSH     (W1H + W2H)              // 114688 halves = 229376 B

// ---------------- prep: swizzle f32 weights into f16 WMMA B-fragments -------
__global__ __launch_bounds__(256) void mmoe_prep_kernel(
    const float* __restrict__ W1, const float* __restrict__ W2,
    _Float16* __restrict__ ws)
{
  int t = blockIdx.x * 256 + threadIdx.x;       // one thread per f16 element
  if (t < W1H) {
    int frag = t >> 9, rem = t & 511;
    int lane = rem >> 4, i = rem & 15;
    int nt = frag & 7, ks = (frag >> 3) % 5, e = frag / 40;
    int half = lane >> 4, lm = lane & 15;
    int k = ks * 32 + half * 16 + i;            // B elem i -> K = 16*half + i
    int n = nt * 16 + lm;                       // B lane  -> N = lane % 16
    float w = (k < DIN) ? W1[((size_t)(e * DIN + k)) * H1 + n] : 0.f;
    ws[t] = (_Float16)w;
  } else if (t < WSH) {
    int t2 = t - W1H;
    int frag = t2 >> 9, rem = t2 & 511;
    int lane = rem >> 4, i = rem & 15;
    int nt = frag & 3, ks = (frag >> 2) & 3, e = frag >> 4;
    int half = lane >> 4, lm = lane & 15;
    int k = ks * 32 + half * 16 + i;
    int n = nt * 16 + lm;
    ws[t] = (_Float16)W2[((size_t)(e * H1 + k)) * H2 + n];
  }
}

// ---------------- fused MMOE forward ----------------------------------------
__global__ __launch_bounds__(256) void mmoe_fused_kernel(
    const int*   __restrict__ sp,    const float* __restrict__ dense,
    const float* __restrict__ emb,
    const _Float16* __restrict__ ws,                       // swizzled W1|W2
    const float* __restrict__ b1,    const float* __restrict__ b2,
    const float* __restrict__ gFW,   const float* __restrict__ gFb,
    const float* __restrict__ gLW,   const float* __restrict__ gLb,
    const float* __restrict__ fW,    const float* __restrict__ fb,
    const float* __restrict__ lW,    const float* __restrict__ lb,
    float* __restrict__ out)
{
  __shared__ alignas(16) _Float16 xs[ROWS2][KPAD];   // 10240 B (x tile, f16)
  // Per-wave 4KB slice, used first as h (f16 [16][128]) then as eo (f32 [16][64]).
  __shared__ alignas(16) unsigned char hseo[8][4096];          // 32768 B
  __shared__ float gw[2][DIN][NEXP];                 // gate weights   4576 B
  __shared__ float lg[2][ROWS2][NEXP];               // gate logits    1024 B
  __shared__ float gates[2][ROWS2][NEXP];            // softmax gates  1024 B

  const int tid  = threadIdx.x;
  const int r0   = blockIdx.x * ROWS2;
  const int w    = tid >> 5;        // wave 0..7
  const int e    = w & 3;           // expert
  const int tile = w >> 2;          // row-tile 0/1
  const int lane = tid & 31;
  const int half = lane >> 4;
  const int lm   = lane & 15;

  // Warm L2 for swizzled weights (global_prefetch_b8), ~1KB per thread.
  for (int off = tid; off < (int)(WSH * 2 / 1024); off += 256)
    __builtin_prefetch((const char*)ws + (size_t)off * 1024, 0, 1);

  // ---- Phase 1: gate weights -> LDS, gather x tile -> LDS (f16) ----
  for (int s = tid; s < DIN * NEXP; s += 256) {
    ((float*)gw[0])[s] = gFW[s];
    ((float*)gw[1])[s] = gLW[s];
  }
  for (int s = tid; s < ROWS2 * NS; s += 256) {
    int r = s / NS, f = s % NS;
    int idx = sp[(size_t)(r0 + r) * NS + f];
    const float* ep = emb + ((size_t)f * VOCAB + (size_t)idx) * EMBD;
    #pragma unroll
    for (int j = 0; j < EMBD; ++j)
      xs[r][ND + f * EMBD + j] = (_Float16)ep[j];
  }
  for (int s = tid; s < ROWS2 * ND; s += 256) {
    int r = s / ND, c = s % ND;
    xs[r][c] = (_Float16)dense[(size_t)(r0 + r) * ND + c];
  }
  for (int s = tid; s < ROWS2 * (KPAD - DIN); s += 256) {
    int r = s / (KPAD - DIN), c = DIN + s % (KPAD - DIN);
    xs[r][c] = (_Float16)0.f;
  }
  __syncthreads();

  // ---- Phase 2: gate logits (1 thread per task,row,expert: 2*32*4 = 256) ----
  {
    int task = tid >> 7;
    int rem  = tid & 127;
    int r = rem >> 2, g = rem & 3;
    float acc = task ? gLb[g] : gFb[g];
    for (int k = 0; k < DIN; ++k)
      acc += (float)xs[r][k] * gw[task][k][g];
    lg[task][r][g] = acc;
  }

  // ---- Phase 3: GEMM1  h = relu(x @ W1[e] + b1[e])  (wave-private) ----
  // A layout (16-bit A 16x32): lane row = L%16; elem i -> K=16*half+i
  // (contiguous, matching B fragments).
  _Float16* hsw = (_Float16*)hseo[w];                // [16][H1]
  half16 ax[5];
  #pragma unroll
  for (int ks = 0; ks < 5; ++ks) {
    int kb = ks * 32 + half * 16;
    half8 lo = *(const half8*)&xs[tile * 16 + lm][kb];
    half8 hi = *(const half8*)&xs[tile * 16 + lm][kb + 8];
    #pragma unroll
    for (int i = 0; i < 8; ++i) { ax[ks][i] = lo[i]; ax[ks][8 + i] = hi[i]; }
  }
  const half16* bw1 = (const half16*)ws;             // fragment table
  #pragma unroll
  for (int nt = 0; nt < 8; ++nt) {
    float8 acc = {};
    #pragma unroll
    for (int ks = 0; ks < 5; ++ks) {
      half16 b = bw1[(size_t)(((e * 5 + ks) << 3) + nt) * 32 + lane];
      acc = __builtin_amdgcn_wmma_f32_16x16x32_f16(false, ax[ks], false, b,
                                                   (short)0, acc, false, false);
    }
    float bias = b1[e * H1 + nt * 16 + lm];
    #pragma unroll
    for (int j = 0; j < 8; ++j) {                    // C: row = j+8*half
      float v = fmaxf(acc[j] + bias, 0.f);
      hsw[(j + 8 * half) * H1 + nt * 16 + lm] = (_Float16)v;
    }
  }
  __syncthreads();                                   // lg visible

  // ---- Phase 4: softmax of gate logits (waves 0-1) ----
  if (tid < 64) {
    int task = tid >> 5, r = tid & 31;
    float l0 = lg[task][r][0], l1 = lg[task][r][1];
    float l2 = lg[task][r][2], l3 = lg[task][r][3];
    float m  = fmaxf(fmaxf(l0, l1), fmaxf(l2, l3));
    float e0 = __expf(l0 - m), e1 = __expf(l1 - m);
    float e2 = __expf(l2 - m), e3 = __expf(l3 - m);
    float inv = 1.f / (e0 + e1 + e2 + e3);
    gates[task][r][0] = e0 * inv; gates[task][r][1] = e1 * inv;
    gates[task][r][2] = e2 * inv; gates[task][r][3] = e3 * inv;
  }

  // ---- Phase 5: GEMM2  eo = relu(h @ W2[e] + b2[e])  (wave-private) ----
  half16 ah[4];
  #pragma unroll
  for (int ks = 0; ks < 4; ++ks) {
    int kb = ks * 32 + half * 16;
    half8 lo = *(const half8*)&hsw[lm * H1 + kb];
    half8 hi = *(const half8*)&hsw[lm * H1 + kb + 8];
    #pragma unroll
    for (int i = 0; i < 8; ++i) { ah[ks][i] = lo[i]; ah[ks][8 + i] = hi[i]; }
  }
  float8 acc2[4];
  const half16* bw2 = (const half16*)(ws + W1H);
  #pragma unroll
  for (int nt = 0; nt < 4; ++nt) {
    float8 acc = {};
    #pragma unroll
    for (int ks = 0; ks < 4; ++ks) {
      half16 b = bw2[(size_t)(((e * 4 + ks) << 2) + nt) * 32 + lane];
      acc = __builtin_amdgcn_wmma_f32_16x16x32_f16(false, ah[ks], false, b,
                                                   (short)0, acc, false, false);
    }
    acc2[nt] = acc;
  }
  // All reads of hsw are done (same-wave LDS is in order) -> overwrite as eo.
  float* eow = (float*)hseo[w];                      // [16][H2]
  #pragma unroll
  for (int nt = 0; nt < 4; ++nt) {
    float bias = b2[e * H2 + nt * 16 + lm];
    #pragma unroll
    for (int j = 0; j < 8; ++j) {
      float v = fmaxf(acc2[nt][j] + bias, 0.f);
      eow[(j + 8 * half) * H2 + nt * 16 + lm] = v;
    }
  }
  __syncthreads();                                   // eo + gates visible

  // ---- Phase 6: gated combine + heads + sigmoid (8 lanes per row) ----
  {
    int r = tid >> 3, i = tid & 7;                   // r in [0,32)
    float gf0 = gates[0][r][0], gf1 = gates[0][r][1];
    float gf2 = gates[0][r][2], gf3 = gates[0][r][3];
    float gl0 = gates[1][r][0], gl1 = gates[1][r][1];
    float gl2 = gates[1][r][2], gl3 = gates[1][r][3];
    int rt = r >> 4, rl = r & 15;
    const float* e0p = (const float*)hseo[rt * 4 + 0] + rl * H2;
    const float* e1p = (const float*)hseo[rt * 4 + 1] + rl * H2;
    const float* e2p = (const float*)hseo[rt * 4 + 2] + rl * H2;
    const float* e3p = (const float*)hseo[rt * 4 + 3] + rl * H2;
    float pf = 0.f, pl = 0.f;
    #pragma unroll
    for (int j = 0; j < 8; ++j) {
      int c = i * 8 + j;
      float v0 = e0p[c], v1 = e1p[c], v2 = e2p[c], v3 = e3p[c];
      float tf = gf0 * v0 + gf1 * v1 + gf2 * v2 + gf3 * v3;
      float tl = gl0 * v0 + gl1 * v1 + gl2 * v2 + gl3 * v3;
      pf += tf * fW[c];
      pl += tl * lW[c];
    }
    pf += __shfl_down(pf, 4, 8); pl += __shfl_down(pl, 4, 8);
    pf += __shfl_down(pf, 2, 8); pl += __shfl_down(pl, 2, 8);
    pf += __shfl_down(pf, 1, 8); pl += __shfl_down(pl, 1, 8);
    if (i == 0) {
      int rg = r0 + r;
      out[rg]         = 1.f / (1.f + __expf(-(pf + fb[0])));   // finish
      out[BATCH + rg] = 1.f / (1.f + __expf(-(pl + lb[0])));   // like
    }
  }
}

extern "C" void kernel_launch(void* const* d_in, const int* in_sizes, int n_in,
                              void* d_out, int out_size, void* d_ws, size_t ws_size,
                              hipStream_t stream) {
  const int*   sp    = (const int*)  d_in[0];
  const float* dense = (const float*)d_in[1];
  const float* emb   = (const float*)d_in[2];
  const float* W1    = (const float*)d_in[3];
  const float* b1    = (const float*)d_in[4];
  const float* W2    = (const float*)d_in[5];
  const float* b2    = (const float*)d_in[6];
  const float* gFW   = (const float*)d_in[7];
  const float* gFb   = (const float*)d_in[8];
  const float* gLW   = (const float*)d_in[9];
  const float* gLb   = (const float*)d_in[10];
  const float* fW    = (const float*)d_in[11];
  const float* fb    = (const float*)d_in[12];
  const float* lW    = (const float*)d_in[13];
  const float* lb    = (const float*)d_in[14];
  float* out = (float*)d_out;
  _Float16* ws = (_Float16*)d_ws;     // needs 229,376 B of scratch

  // 1) swizzle weights into WMMA B-fragment layout (f16)
  mmoe_prep_kernel<<<dim3((WSH + 255) / 256), dim3(256), 0, stream>>>(W1, W2, ws);
  // 2) fused forward: 2048 blocks x 8 waves (expert x row-tile)
  mmoe_fused_kernel<<<dim3(BATCH / ROWS2), dim3(256), 0, stream>>>(
      sp, dense, emb, ws, b1, b2, gFW, gFb, gLW, gLb, fW, fb, lW, lb, out);
}
